// MambaBlock_v2_1520418422823
// MI455X (gfx1250) — compile-verified, matrix operand order checked by reading, not run on hardware
//
#include <hip/hip_runtime.h>
#include <cstdint>

// ---------------- Mamba2 block for MI455X (gfx1250), wave32 + WMMA bf16 ----
// B=8, L=1024, D_MODEL=256, D_INNER=512, NHEADS=16, HEADDIM=32, D_STATE=128
// D_CONV=4, CONV_DIM=768, D_IN_PROJ=1296, CHUNK=256 (4 chunks), GN_GROUPS=8

#define DEV __device__ __forceinline__

typedef __attribute__((ext_vector_type(16))) __bf16 bf16x16;
typedef __attribute__((ext_vector_type(8)))  __bf16 bf16x8;
typedef __attribute__((ext_vector_type(8)))  float  f32x8;

DEV float sigf(float x)  { return 1.0f / (1.0f + expf(-x)); }
DEV float siluf(float x) { return x * sigf(x); }
DEV float softplusf(float x) { return (x > 20.0f) ? x : log1pf(expf(x)); }

// A/B fragment loader for v_wmma_f32_16x16x32_bf16.
// Source is row-major [row][k] with row stride `ld` (elements, multiple of 8).
// Per ISA 16-bit A 16x32 layout: lanes 0-15 hold K {k0..k0+7, k0+16..k0+23},
// lanes 16-31 hold K {k0+8..k0+15, k0+24..k0+31}; row = row0 + (lane&15).
// B fragment is the mirror (lane&15 = N column) reading [n][k] storage.
DEV bf16x16 load_ab_frag(const __bf16* base, int row0, int ld, int k0) {
  const int lane = threadIdx.x & 31;
  const int row  = row0 + (lane & 15);
  const int hi   = lane >> 4;
  const __bf16* p = base + (size_t)row * ld + k0 + hi * 8;
  bf16x8 lo = *reinterpret_cast<const bf16x8*>(p);
  bf16x8 hh = *reinterpret_cast<const bf16x8*>(p + 16);
  bf16x16 f;
#pragma unroll
  for (int i = 0; i < 8; ++i) { f[i] = lo[i]; f[i + 8] = hh[i]; }
  return f;
}

DEV f32x8 wmma_bf16(bf16x16 a, bf16x16 b, f32x8 c) {
  return __builtin_amdgcn_wmma_f32_16x16x32_bf16(false, a, false, b,
                                                 (short)0, c, false, false);
}

// C/D layout: element v -> (M = v + 8*(lane>>4), N = lane&15)
DEV void store_cd_frag(float* base, int row0, int ld, int col0, f32x8 c) {
  const int lane = threadIdx.x & 31;
  const int hi = lane >> 4, n = lane & 15;
#pragma unroll
  for (int v = 0; v < 8; ++v)
    base[(size_t)(row0 + v + 8 * hi) * ld + col0 + n] = c[v];
}

// ---------------------------------------------------------------- staging --
__global__ void k_stage_inputs(const float* __restrict__ x,
                               const float* __restrict__ W_in,
                               const float* __restrict__ W_out,
                               __bf16* __restrict__ u16,
                               __bf16* __restrict__ Win16,
                               __bf16* __restrict__ Wout16) {
  int idx = blockIdx.x * blockDim.x + threadIdx.x;
  const int NU = 8 * 1024 * 256;   // u[b][l][ci] = x[b][ci][l]
  const int NW1 = 1296 * 256;
  const int NW2 = 256 * 512;
  if (idx < NU) {
    int ci = idx & 255, l = (idx >> 8) & 1023, b = idx >> 18;
    u16[idx] = (__bf16)x[((size_t)(b * 256 + ci)) * 1024 + l];
  }
  if (idx < NW1) Win16[idx] = (__bf16)W_in[idx];
  if (idx < NW2) Wout16[idx] = (__bf16)W_out[idx];
}

// ----------------------------------------------------------- in-proj GEMM --
// Z[8192][1296] = U[8192x256] @ W_in^T  (W_in: 1296x256 row-major)
__global__ void __launch_bounds__(128) k_gemm_inproj(const __bf16* __restrict__ U,
                                                     const __bf16* __restrict__ W,
                                                     float* __restrict__ Z) {
  const int wave = threadIdx.x >> 5;
  const int mt = blockIdx.y * 4 + wave;  // 0..511
  const int nt = blockIdx.x;             // 0..80
  f32x8 acc = {};
#pragma unroll
  for (int k0 = 0; k0 < 256; k0 += 32) {
    bf16x16 a = load_ab_frag(U, mt * 16, 256, k0);
    bf16x16 b = load_ab_frag(W, nt * 16, 256, k0);
    acc = wmma_bf16(a, b, acc);
  }
  store_cd_frag(Z, mt * 16, 1296, nt * 16, acc);
}

// ------------------------------------------------- depthwise conv + SiLU ---
__global__ void k_conv(const float* __restrict__ zx,
                       const float* __restrict__ conv_w,
                       const float* __restrict__ conv_b,
                       float* __restrict__ xconv) {
  int idx = blockIdx.x * blockDim.x + threadIdx.x;
  if (idx >= 8 * 1024 * 768) return;
  int c = idx % 768, l = (idx / 768) % 1024, b = idx / (768 * 1024);
  float acc = conv_b[c];
#pragma unroll
  for (int k = 0; k < 4; ++k) {
    int ls = l + k - 3;
    float xv = (ls >= 0) ? zx[((size_t)(b * 1024 + ls)) * 1296 + 512 + c] : 0.0f;
    acc += conv_w[c * 4 + k] * xv;
  }
  xconv[((size_t)(b * 1024 + l)) * 768 + c] = siluf(acc);
}

// ------------------------------------- dt softplus + per-chunk cumsum(A*dt) -
__global__ void k_dt_scan(const float* __restrict__ zx,
                          const float* __restrict__ dt_bias,
                          const float* __restrict__ A_log,
                          float* __restrict__ dtsp, float* __restrict__ acs) {
  int t = blockIdx.x * blockDim.x + threadIdx.x;
  if (t >= 128) return;
  int h = t & 15, b = t >> 4;
  float Ah = -expf(A_log[h]);
  float bias = dt_bias[h];
  for (int c = 0; c < 4; ++c) {
    float run = 0.0f;
    for (int s = 0; s < 256; ++s) {
      int l = c * 256 + s;
      float d = softplusf(zx[((size_t)(b * 1024 + l)) * 1296 + 1280 + h] + bias);
      dtsp[(b * 1024 + l) * 16 + h] = d;
      run += d * Ah;
      acs[(((b * 4 + c) * 16 + h)) * 256 + s] = run;  // inclusive cumsum
    }
  }
}

// ------------------------------------------------ stage B/C chunks as bf16 -
// Bb[bc][s][n], Bt[bc][n][s], Cb[bc][s][n]
__global__ void k_stage_bc(const float* __restrict__ xconv,
                           __bf16* __restrict__ Bb, __bf16* __restrict__ Bt,
                           __bf16* __restrict__ Cb) {
  int idx = blockIdx.x * blockDim.x + threadIdx.x;
  if (idx >= 8 * 4 * 256 * 128) return;
  int n = idx & 127, s = (idx >> 7) & 255, bc = idx >> 15;
  int l = (bc & 3) * 256 + s, b = bc >> 2;
  float bv = xconv[((size_t)(b * 1024 + l)) * 768 + 512 + n];
  float cv = xconv[((size_t)(b * 1024 + l)) * 768 + 640 + n];
  Bb[((size_t)bc * 256 + s) * 128 + n] = (__bf16)bv;
  Bt[((size_t)bc * 128 + n) * 256 + s] = (__bf16)bv;
  Cb[((size_t)bc * 256 + s) * 128 + n] = (__bf16)cv;
}

// ----------------------- stage X^T (dt*xs) and decay-scaled X^T as bf16 ----
// Xt[bch][p][s], Xdec[bch][p][s] = Xt * exp(acs_last - acs[s])
__global__ void k_stage_x(const float* __restrict__ xconv,
                          const float* __restrict__ dtsp,
                          const float* __restrict__ acs,
                          __bf16* __restrict__ Xt, __bf16* __restrict__ Xdec) {
  int idx = blockIdx.x * blockDim.x + threadIdx.x;
  if (idx >= 512 * 32 * 256) return;
  int s = idx & 255, p = (idx >> 8) & 31, bch = idx >> 13;
  int h = bch & 15, bc = bch >> 4, b = bc >> 2, c = bc & 3;
  int l = c * 256 + s;
  float xv = xconv[((size_t)(b * 1024 + l)) * 768 + h * 32 + p];
  float d  = dtsp[(b * 1024 + l) * 16 + h];
  float xd = d * xv;
  float decay = expf(acs[(size_t)bch * 256 + 255] - acs[(size_t)bch * 256 + s]);
  Xt[((size_t)bch * 32 + p) * 256 + s]   = (__bf16)xd;
  Xdec[((size_t)bch * 32 + p) * 256 + s] = (__bf16)(xd * decay);
}

// ------------------------------- chunk states: S[p][n] = (decay*X)^T @ B ---
__global__ void __launch_bounds__(512) k_states(const __bf16* __restrict__ Xdec,
                                                const __bf16* __restrict__ Bt,
                                                float* __restrict__ states) {
  int bch = blockIdx.x;        // 512 = (b*4+c)*16+h
  int bc = bch >> 4;
  int wave = threadIdx.x >> 5; // 16 waves: mt (p tile) x nt (n tile)
  int mt = wave & 1, nt = wave >> 1;
  const __bf16* A  = Xdec + (size_t)bch * 32 * 256;
  const __bf16* Bm = Bt   + (size_t)bc * 128 * 256;
  f32x8 acc = {};
#pragma unroll
  for (int k0 = 0; k0 < 256; k0 += 32) {
    bf16x16 a = load_ab_frag(A,  mt * 16, 256, k0);
    bf16x16 b = load_ab_frag(Bm, nt * 16, 256, k0);
    acc = wmma_bf16(a, b, acc);
  }
  store_cd_frag(states + (size_t)bch * 32 * 128, mt * 16, 128, nt * 16, acc);
}

// ------------------- inter-chunk recurrence: P_z = e^{Acl_{z-1}} P + S ----
__global__ void k_chunkscan(const float* __restrict__ states,
                            const float* __restrict__ acs,
                            __bf16* __restrict__ prevb) {
  int idx = blockIdx.x * blockDim.x + threadIdx.x;
  if (idx >= 8 * 16 * 32 * 128) return;
  int n = idx & 127, p = (idx >> 7) & 31, h = (idx >> 12) & 15, b = idx >> 16;
  float P = 0.0f;
  for (int z = 0; z < 4; ++z) {
    if (z > 0) {
      int pc = (b * 4 + (z - 1)) * 16 + h;
      float Acl = acs[(size_t)pc * 256 + 255];
      P = expf(Acl) * P + states[((size_t)pc * 32 + p) * 128 + n];
    }
    int bch = (b * 4 + z) * 16 + h;
    prevb[((size_t)bch * 32 + p) * 128 + n] = (__bf16)P;
  }
}

// --------------------------- main SSD kernel: Y = Ydiag + Yoff + D*xs -----
__global__ void __launch_bounds__(256) k_ssd_y(
    const __bf16* __restrict__ Cb, const __bf16* __restrict__ Bb,
    const __bf16* __restrict__ Xt, const __bf16* __restrict__ prevb,
    const float* __restrict__ acs, const float* __restrict__ xconv,
    const float* __restrict__ Dp, float* __restrict__ Ybuf) {
  __shared__ float acs_sh[256];
  __shared__ __align__(16) __bf16 Gs[8][16][32];  // per-wave G relayout tile
  const int bch = blockIdx.x;
  const int h = bch & 15, bc = bch >> 4, b = bc >> 2, c = bc & 3;
  const int wave = threadIdx.x >> 5, lane = threadIdx.x & 31;
  const int hi = lane >> 4, nn = lane & 15;
  acs_sh[threadIdx.x] = acs[(size_t)bch * 256 + threadIdx.x];
  __syncthreads();
  const __bf16* Cc = Cb + (size_t)bc * 256 * 128;
  const __bf16* Bc = Bb + (size_t)bc * 256 * 128;
  const __bf16* Xc = Xt + (size_t)bch * 32 * 256;
  const __bf16* Pv = prevb + (size_t)bch * 32 * 128;
  const float dph = Dp[h];
  for (int i = 0; i < 2; ++i) {
    const int lt = wave * 2 + i;  // l-tile 0..15 (uniform per wave)
    f32x8 y0 = {}, y1 = {}, o0 = {}, o1 = {};
    // Y_off = C @ prev^T (K = n = 128)
#pragma unroll
    for (int k0 = 0; k0 < 128; k0 += 32) {
      bf16x16 a  = load_ab_frag(Cc, lt * 16, 128, k0);
      bf16x16 b0 = load_ab_frag(Pv, 0,  128, k0);
      bf16x16 b1 = load_ab_frag(Pv, 16, 128, k0);
      o0 = wmma_bf16(a, b0, o0);
      o1 = wmma_bf16(a, b1, o1);
    }
    // Y_diag: s-tile pairs up to the diagonal
    for (int st2 = 0; st2 <= (lt >> 1); ++st2) {
#pragma unroll
      for (int half = 0; half < 2; ++half) {
        const int st = st2 * 2 + half;
        f32x8 g = {};
#pragma unroll
        for (int k0 = 0; k0 < 128; k0 += 32) {
          bf16x16 a = load_ab_frag(Cc, lt * 16, 128, k0);
          bf16x16 b = load_ab_frag(Bc, st * 16, 128, k0);
          g = wmma_bf16(a, b, g);  // G[l][s] = sum_n C[l][n]B[s][n]
        }
        const int s = st * 16 + nn;
#pragma unroll
        for (int v = 0; v < 8; ++v) {
          const int lrow = v + 8 * hi;
          const int l = lt * 16 + lrow;
          float f = (s <= l) ? expf(acs_sh[l] - acs_sh[s]) : 0.0f;
          Gs[wave][lrow][half * 16 + nn] = (__bf16)(g[v] * f);
        }
      }
      // (G masked) @ X : A from LDS relayout, B from Xt[p][s]
      bf16x16 ga  = load_ab_frag(&Gs[wave][0][0], 0, 32, 0);
      bf16x16 xb0 = load_ab_frag(Xc, 0,  256, st2 * 32);
      bf16x16 xb1 = load_ab_frag(Xc, 16, 256, st2 * 32);
      y0 = wmma_bf16(ga, xb0, y0);
      y1 = wmma_bf16(ga, xb1, y1);
    }
    // combine + skip term, store Y[b][l][h*32+p]
#pragma unroll
    for (int v = 0; v < 8; ++v) {
      const int l = lt * 16 + v + 8 * hi;
      const float el = expf(acs_sh[l]);
      const int lg = c * 256 + l;
      const float xs0 = xconv[((size_t)(b * 1024 + lg)) * 768 + h * 32 + nn];
      const float xs1 = xconv[((size_t)(b * 1024 + lg)) * 768 + h * 32 + 16 + nn];
      float* yr = Ybuf + ((size_t)(b * 1024 + lg)) * 512 + h * 32;
      yr[nn]      = y0[v] + el * o0[v] + dph * xs0;
      yr[16 + nn] = y1[v] + el * o1[v] + dph * xs1;
    }
  }
}

// ----------------------------------------- gate with silu(z) + RMSNorm ----
__global__ void __launch_bounds__(256) k_gate_rms(const float* __restrict__ Ybuf,
                                                  const float* __restrict__ zx,
                                                  const float* __restrict__ rms_w,
                                                  __bf16* __restrict__ yn) {
  __shared__ float red[256];
  const int row = blockIdx.x;  // b*1024 + l
  const int t = threadIdx.x;
  float v0 = Ybuf[(size_t)row * 512 + t]       * siluf(zx[(size_t)row * 1296 + t]);
  float v1 = Ybuf[(size_t)row * 512 + 256 + t] * siluf(zx[(size_t)row * 1296 + 256 + t]);
  red[t] = v0 * v0 + v1 * v1;
  __syncthreads();
  for (int off = 128; off > 0; off >>= 1) {
    if (t < off) red[t] += red[t + off];
    __syncthreads();
  }
  const float rstd = rsqrtf(red[0] / 512.0f + 1e-5f);
  yn[(size_t)row * 512 + t]       = (__bf16)(v0 * rstd * rms_w[t]);
  yn[(size_t)row * 512 + 256 + t] = (__bf16)(v1 * rstd * rms_w[256 + t]);
}

// ------------------------------------------------------------ out GEMM ----
// out_pre[b][m][l] = sum_d yn[b*1024+l][d] * W_out[m][d]
__global__ void __launch_bounds__(128) k_gemm_out(const __bf16* __restrict__ yn,
                                                  const __bf16* __restrict__ W,
                                                  float* __restrict__ out_pre) {
  const int wave = threadIdx.x >> 5;
  const int mt = blockIdx.y * 4 + wave;  // 0..511 (rows b*1024+l)
  const int nt = blockIdx.x;             // 0..15  (m tiles)
  f32x8 acc = {};
#pragma unroll
  for (int k0 = 0; k0 < 512; k0 += 32) {
    bf16x16 a = load_ab_frag(yn, mt * 16, 512, k0);
    bf16x16 b = load_ab_frag(W,  nt * 16, 512, k0);
    acc = wmma_bf16(a, b, acc);
  }
  const int lane = threadIdx.x & 31;
  const int hi = lane >> 4, n = lane & 15;
  const int m = nt * 16 + n;
#pragma unroll
  for (int v = 0; v < 8; ++v) {
    int row = mt * 16 + v + 8 * hi;
    int b = row >> 10, l = row & 1023;
    out_pre[((size_t)(b * 256 + m)) * 1024 + l] = acc[v];
  }
}

// --------------------------------------------------------- group norm -----
__global__ void __launch_bounds__(256) k_gn_stats(const float* __restrict__ out_pre,
                                                  float* __restrict__ stats) {
  __shared__ float s1[256], s2[256];
  const int bg = blockIdx.x;  // b*8+g, 64 groups
  const int b = bg >> 3, g = bg & 7;
  const float* base = out_pre + ((size_t)(b * 256 + g * 32)) * 1024;
  float a = 0.0f, q = 0.0f;
  for (int i = threadIdx.x; i < 32 * 1024; i += 256) {
    float v = base[i]; a += v; q += v * v;
  }
  s1[threadIdx.x] = a; s2[threadIdx.x] = q;
  __syncthreads();
  for (int off = 128; off > 0; off >>= 1) {
    if (threadIdx.x < off) {
      s1[threadIdx.x] += s1[threadIdx.x + off];
      s2[threadIdx.x] += s2[threadIdx.x + off];
    }
    __syncthreads();
  }
  if (threadIdx.x == 0) {
    float mu = s1[0] / 32768.0f;
    float var = s2[0] / 32768.0f - mu * mu;
    stats[bg * 2] = mu;
    stats[bg * 2 + 1] = rsqrtf(var + 1e-5f);
  }
}

__global__ void k_gn_apply(const float* __restrict__ out_pre,
                           const float* __restrict__ stats,
                           const float* __restrict__ gn_w,
                           const float* __restrict__ gn_b,
                           float* __restrict__ out) {
  int idx = blockIdx.x * blockDim.x + threadIdx.x;
  if (idx >= 8 * 256 * 1024) return;
  int m = (idx >> 10) & 255, b = idx >> 18;
  int bg = b * 8 + (m >> 5);
  float v = (out_pre[idx] - stats[bg * 2]) * stats[bg * 2 + 1] * gn_w[m] + gn_b[m];
  out[idx] = v * tanhf(softplusf(v));  // mish
}

// ----------------------------------------------------------------- host ---
extern "C" void kernel_launch(void* const* d_in, const int* in_sizes, int n_in,
                              void* d_out, int out_size, void* d_ws, size_t ws_size,
                              hipStream_t stream) {
  const float* x       = (const float*)d_in[0];
  const float* W_in    = (const float*)d_in[1];
  const float* conv_w  = (const float*)d_in[2];
  const float* conv_b  = (const float*)d_in[3];
  const float* dt_bias = (const float*)d_in[4];
  const float* A_log   = (const float*)d_in[5];
  const float* Dp      = (const float*)d_in[6];
  const float* rms_w   = (const float*)d_in[7];
  const float* W_out   = (const float*)d_in[8];
  const float* gn_w    = (const float*)d_in[9];
  const float* gn_b    = (const float*)d_in[10];
  float* out = (float*)d_out;

  char* ws = (char*)d_ws;
  size_t off = 0;
  auto alloc = [&](size_t bytes) {
    size_t r = off;
    off = (off + bytes + 255) & ~(size_t)255;
    return r;
  };
  float*  zx      = (float*) (ws + alloc((size_t)8192 * 1296 * 4));
  float*  xconv   = (float*) (ws + alloc((size_t)8192 * 768 * 4));
  float*  dtsp    = (float*) (ws + alloc((size_t)8192 * 16 * 4));
  float*  acs     = (float*) (ws + alloc((size_t)512 * 256 * 4));
  __bf16* Xt      = (__bf16*)(ws + alloc((size_t)512 * 32 * 256 * 2));
  __bf16* Xdec    = (__bf16*)(ws + alloc((size_t)512 * 32 * 256 * 2));
  __bf16* Bb16    = (__bf16*)(ws + alloc((size_t)32 * 256 * 128 * 2));
  __bf16* Bt16    = (__bf16*)(ws + alloc((size_t)32 * 128 * 256 * 2));
  __bf16* Cb16    = (__bf16*)(ws + alloc((size_t)32 * 256 * 128 * 2));
  float*  states  = (float*) (ws + alloc((size_t)512 * 32 * 128 * 4));
  __bf16* prevb16 = (__bf16*)(ws + alloc((size_t)512 * 32 * 128 * 2));
  float*  Ybuf    = (float*) (ws + alloc((size_t)8192 * 512 * 4));
  __bf16* ynb16   = (__bf16*)(ws + alloc((size_t)8192 * 512 * 2));
  float*  out_pre = (float*) (ws + alloc((size_t)8192 * 256 * 4));
  __bf16* u16     = (__bf16*)(ws + alloc((size_t)8192 * 256 * 2));
  __bf16* Win16   = (__bf16*)(ws + alloc((size_t)1296 * 256 * 2));
  __bf16* Wout16  = (__bf16*)(ws + alloc((size_t)256 * 512 * 2));
  float*  stats   = (float*) (ws + alloc((size_t)64 * 2 * 4));
  (void)ws_size; (void)in_sizes; (void)n_in; (void)out_size;

  k_stage_inputs<<<8192, 256, 0, stream>>>(x, W_in, W_out, u16, Win16, Wout16);
  k_gemm_inproj<<<dim3(81, 128), 128, 0, stream>>>(u16, Win16, zx);
  k_conv<<<24576, 256, 0, stream>>>(zx, conv_w, conv_b, xconv);
  k_dt_scan<<<1, 128, 0, stream>>>(zx, dt_bias, A_log, dtsp, acs);
  k_stage_bc<<<4096, 256, 0, stream>>>(xconv, Bb16, Bt16, Cb16);
  k_stage_x<<<16384, 256, 0, stream>>>(xconv, dtsp, acs, Xt, Xdec);
  k_states<<<512, 512, 0, stream>>>(Xdec, Bt16, states);
  k_chunkscan<<<2048, 256, 0, stream>>>(states, acs, prevb16);
  k_ssd_y<<<512, 256, 0, stream>>>(Cb16, Bb16, Xt, prevb16, acs, xconv, Dp, Ybuf);
  k_gate_rms<<<8192, 256, 0, stream>>>(Ybuf, zx, rms_w, ynb16);
  k_gemm_out<<<dim3(16, 128), 128, 0, stream>>>(ynb16, Wout16, out_pre);
  k_gn_stats<<<64, 256, 0, stream>>>(out_pre, stats);
  k_gn_apply<<<8192, 256, 0, stream>>>(out_pre, stats, gn_w, gn_b, out);
}
